// TokenGridCorrFiLMS4_84963043049920
// MI455X (gfx1250) — matrix operand order repaired and checked by reading, not compiled, not run on hardware
//
#include <hip/hip_runtime.h>
#include <math.h>
#include <stdint.h>

// ---------------- problem constants ----------------
#define B_     16
#define T_     512
#define DT_    1024
#define G_     576
#define DV_    768
#define H_     4
#define DH_    32
#define INNER_ 128
#define HID_   4096
#define NST_   64
#define KTOP_  32

typedef _Float16 v16h __attribute__((ext_vector_type(16)));
typedef _Float16 v8h  __attribute__((ext_vector_type(8)));
typedef float    v8f  __attribute__((ext_vector_type(8)));

// =========================================================
// f16 WMMA GEMM:  C[M,N] = alpha * A[M,K] * Bt[N,K]^T
// A, Bt f16 row-major.  LDS staging double-buffered, one barrier per K-step.
// Staging engine: Tensor Data Mover (tensor_load_to_lds, one DMA per tile,
// descriptor pad fields reproduce the 40-half padded LDS pitch) when the
// builtin exists; otherwise per-lane global_load_async_to_lds_b128.
// =========================================================

struct GPH {
    const _Float16* A; const _Float16* Bt;
    float* C; _Float16* Ch; _Float16* Cht;
    const float* bias; const float* resid;
    const unsigned char* rmask;                  // optional per-row mask (zero out row)
    int M, N, K, lda, ldb, ldc, ldct, RB, mode;  // mode 0=store 1=+bias 2=+bias,gelu 3=(+bias)+resid
    long long sAb, sBb, sCb, sCtb, sRm;
    float alpha;
};

__device__ __forceinline__ uint32_t lds_off(const void* p) {
    // generic pointer to LDS: low 32 bits are the LDS byte offset
    return (uint32_t)(uintptr_t)p;
}

__device__ __forceinline__ void async_cp16(const _Float16* g, uint32_t lds) {
    // 16-byte global -> LDS async copy (ASYNCcnt-tracked)
    asm volatile("global_load_async_to_lds_b128 %0, %1, off"
                 :: "v"(lds), "v"(g) : "memory");
}

__device__ __forceinline__ void wait_async0() {
    asm volatile("s_wait_asynccnt 0x0" ::: "memory");
}

#if __has_builtin(__builtin_amdgcn_tensor_load_to_lds)
#define HAVE_TDM 1
typedef unsigned int u32x4 __attribute__((ext_vector_type(4)));
typedef int i32x4 __attribute__((ext_vector_type(4)));
typedef int i32x8 __attribute__((ext_vector_type(8)));

// One TDM DMA: rows x 32 f16 tile, row stride lda (elements), into padded LDS
// (row pitch 40 halves = 64B data + 16B pad -> pad_interval=16 DW, pad_amount=4 DW).
__device__ __forceinline__ void tdm_load_tile(const _Float16* g, uint32_t lds,
                                              int rows, int lda) {
    uint64_t ga = (uint64_t)(uintptr_t)g;
    u32x4 g0;
    g0[0] = 1u;                                               // count=1 (valid user D#)
    g0[1] = lds;                                              // lds_addr
    g0[2] = (uint32_t)ga;                                     // global_addr[31:0]
    g0[3] = (uint32_t)((ga >> 32) & 0x01FFFFFFu) | (2u << 30);// global_addr[56:32] | type=2
    i32x8 g1;
    g1[0] = (1 << 16)      // data_size = 2 bytes
          | (1 << 20)      // pad_enable
          | (3 << 22)      // pad_interval: 16 DWORDs (64B)
          | (3 << 25);     // pad_amount: 4 DWORDs (16B)
    g1[1] = (32 & 0xFFFF) << 16;          // tensor_dim0[15:0] = 32  (bits 63:48)
    g1[2] = (rows & 0xFFFF) << 16;        // dim0[31:16]=0 | tensor_dim1[15:0]=rows
    g1[3] = (32 << 16);                   // dim1[31:16]=0 | tile_dim0 = 32
    g1[4] = rows & 0xFFFF;                // tile_dim1 = rows, tile_dim2 = 0
    g1[5] = lda;                          // tensor_dim0_stride[31:0]
    g1[6] = 0;                            // stride0[47:32] | stride1[15:0]
    g1[7] = 0;
    i32x4 z4 = {0, 0, 0, 0};
    i32x8 z8 = {0, 0, 0, 0, 0, 0, 0, 0};
    __builtin_amdgcn_tensor_load_to_lds(g0, g1, z4, z4, z8, 0);  // 6-arg (clang-23) form
}
__device__ __forceinline__ void tdm_wait0() { __builtin_amdgcn_s_wait_tensorcnt(0); }
#else
#define HAVE_TDM 0
#endif

__device__ __forceinline__ v16h frag_a(const _Float16 (*s)[40], int base, int lane) {
    int row = base + (lane & 15);
    int kh  = (lane >> 4) << 3;              // 0 or 8
    v8h lo = *(const v8h*)&s[row][kh];
    v8h hi = *(const v8h*)&s[row][16 + kh];
    v16h r;
#pragma unroll
    for (int i = 0; i < 8; ++i) { r[i] = lo[i]; r[i + 8] = hi[i]; }
    return r;
}

__device__ __forceinline__ v16h frag_b(const _Float16 (*s)[40], int base, int lane) {
    int n  = base + (lane & 15);
    int k0 = (lane >> 4) << 3;               // 0 or 8
    v8h lo = *(const v8h*)&s[n][k0];
    v8h hi = *(const v8h*)&s[n][16 + k0];
    v16h r;
#pragma unroll
    for (int i = 0; i < 8; ++i) { r[i] = lo[i]; r[i + 8] = hi[i]; }
    return r;
}

template <int BM, int BN, int WM, int WN, int MI, int NI, int USE_TDM>
__global__ __launch_bounds__(WM * WN * 32) void gemm_h(GPH p) {
    __shared__ _Float16 sA[2][BM][40];
    __shared__ _Float16 sB[2][BN][40];

    const int tid  = threadIdx.x;
    const int lane = tid & 31;
    const int wave = tid >> 5;
    const int wm   = wave / WN;
    const int wn   = wave % WN;

    const int z = blockIdx.z;
    const _Float16* A  = p.A  + (size_t)z * p.sAb;
    const _Float16* Bt = p.Bt + (size_t)z * p.sBb;

    const int bm = blockIdx.y * BM;
    const int bn = blockIdx.x * BN;

    v8f acc[MI][NI];
#pragma unroll
    for (int i = 0; i < MI; ++i)
#pragma unroll
        for (int j = 0; j < NI; ++j)
#pragma unroll
            for (int e = 0; e < 8; ++e) acc[i][j][e] = 0.0f;

    // per-lane staging geometry (async fallback): THREADS == BM*2 == BN*2
    const int r  = tid >> 1;
    const int hf = tid & 1;
    const _Float16* agp = A  + (size_t)(bm + r) * p.lda + hf * 16;
    const _Float16* bgp = Bt + (size_t)(bn + r) * p.ldb + hf * 16;
    const uint32_t laA[2] = { lds_off(&sA[0][r][hf * 16]), lds_off(&sA[1][r][hf * 16]) };
    const uint32_t laB[2] = { lds_off(&sB[0][r][hf * 16]), lds_off(&sB[1][r][hf * 16]) };
    // tile-base LDS addresses (TDM path)
    const uint32_t tlA[2] = { lds_off(&sA[0][0][0]), lds_off(&sA[1][0][0]) };
    const uint32_t tlB[2] = { lds_off(&sB[0][0][0]), lds_off(&sB[1][0][0]) };
    (void)tlA; (void)tlB;

    // ---- prologue: stage first K tile into buffer 0 ----
#if HAVE_TDM
    if (USE_TDM) {
        if (wave == 0) {
            tdm_load_tile(A  + (size_t)bm * p.lda, tlA[0], BM, p.lda);
            tdm_load_tile(Bt + (size_t)bn * p.ldb, tlB[0], BN, p.ldb);
        }
    } else
#endif
    {
        async_cp16(agp,     laA[0]);  async_cp16(agp + 8, laA[0] + 16);
        async_cp16(bgp,     laB[0]);  async_cp16(bgp + 8, laB[0] + 16);
    }

    int buf = 0;
    for (int kk = 0; kk < p.K; kk += 32) {
        // wait for copies into current buffer
#if HAVE_TDM
        if (USE_TDM) {
            if (wave == 0) tdm_wait0();
        } else
#endif
        {
            wait_async0();
        }
        __syncthreads();                       // buffer visible to all; old reads retired

        const int nxt = kk + 32;
        if (nxt < p.K) {                       // stage next tile into other buffer
#if HAVE_TDM
            if (USE_TDM) {
                if (wave == 0) {
                    tdm_load_tile(A  + (size_t)bm * p.lda + nxt, tlA[buf ^ 1], BM, p.lda);
                    tdm_load_tile(Bt + (size_t)bn * p.ldb + nxt, tlB[buf ^ 1], BN, p.ldb);
                }
            } else
#endif
            {
                async_cp16(agp + nxt,     laA[buf ^ 1]);
                async_cp16(agp + nxt + 8, laA[buf ^ 1] + 16);
                async_cp16(bgp + nxt,     laB[buf ^ 1]);
                async_cp16(bgp + nxt + 8, laB[buf ^ 1] + 16);
                if (nxt + 32 < p.K) {
                    __builtin_prefetch(agp + nxt + 32, 0, 3);
                    __builtin_prefetch(bgp + nxt + 32, 0, 3);
                }
            }
        }

        v16h af[MI], bf[NI];
#pragma unroll
        for (int mi = 0; mi < MI; ++mi) af[mi] = frag_a(sA[buf], (wm * MI + mi) * 16, lane);
#pragma unroll
        for (int ni = 0; ni < NI; ++ni) bf[ni] = frag_b(sB[buf], (wn * NI + ni) * 16, lane);
#pragma unroll
        for (int mi = 0; mi < MI; ++mi)
#pragma unroll
            for (int ni = 0; ni < NI; ++ni)
                acc[mi][ni] = __builtin_amdgcn_wmma_f32_16x16x32_f16(
                    false, af[mi], false, bf[ni], (short)0, acc[mi][ni], false, false);
        buf ^= 1;
    }

    // epilogue — C layout: VGPR e: lanes0-15 -> M=e, lanes16-31 -> M=e+8; N = lane&15
    const int ln = lane & 15;
    const int mh = (lane >> 4) * 8;
    float* Cz = p.C ? p.C + (size_t)z * p.sCb : nullptr;
    _Float16* Chz = p.Ch ? p.Ch + (size_t)z * p.sCb : nullptr;
    const float* Rs = p.resid ? p.resid + (size_t)z * p.sCb : nullptr;
#pragma unroll
    for (int mi = 0; mi < MI; ++mi) {
#pragma unroll
        for (int ni = 0; ni < NI; ++ni) {
            int col = bn + (wn * NI + ni) * 16 + ln;
#pragma unroll
            for (int e = 0; e < 8; ++e) {
                int row = bm + (wm * MI + mi) * 16 + mh + e;
                float v = acc[mi][ni][e] * p.alpha;
                if (p.mode == 1) {
                    v += p.bias[col];
                } else if (p.mode == 2) {
                    v += p.bias[col];
                    v = 0.5f * v * (1.0f + erff(v * 0.70710678118654752f));  // exact gelu
                } else if (p.mode == 3) {
                    if (p.bias) v += p.bias[col];
                    v += Rs[(size_t)row * p.ldc + col];
                }
                if (p.rmask && !p.rmask[(size_t)z * p.sRm + row]) v = 0.0f;
                if (Cz)  Cz [(size_t)row * p.ldc + col] = v;
                if (Chz) Chz[(size_t)row * p.ldc + col] = (_Float16)v;
                if (p.Cht) {
                    int bb = row / p.RB, rr = row - bb * p.RB;
                    p.Cht[(size_t)bb * p.sCtb + (size_t)col * p.ldct + rr] = (_Float16)v;
                }
            }
        }
    }
}

// ================= conversion kernels =================
__global__ __launch_bounds__(256) void cvt16(const float* __restrict__ s,
                                             _Float16* __restrict__ d, size_t n) {
    size_t i = (size_t)blockIdx.x * 256 + threadIdx.x;
    if (i < n) d[i] = (_Float16)s[i];
}

// dst[N][K] = (f16) src[K][N]
__global__ __launch_bounds__(256) void transpose_cvt(const float* __restrict__ src,
                                                     _Float16* __restrict__ dst,
                                                     int K, int N) {
    __shared__ _Float16 tile[32][33];
    int k0 = blockIdx.y * 32, n0 = blockIdx.x * 32;
    int r = threadIdx.x >> 5, c = threadIdx.x & 31;
#pragma unroll
    for (int i = 0; i < 4; ++i)
        tile[r + i * 8][c] = (_Float16)src[(size_t)(k0 + r + i * 8) * N + n0 + c];
    __syncthreads();
#pragma unroll
    for (int i = 0; i < 4; ++i)
        dst[(size_t)(n0 + r + i * 8) * K + k0 + c] = tile[c][r + i * 8];
}

// ================= LayerNorm (one block per row) =================
__global__ __launch_bounds__(256) void ln_kernel(const float* __restrict__ x,
                                                 const float* __restrict__ g,
                                                 const float* __restrict__ b,
                                                 float* __restrict__ out32,
                                                 _Float16* __restrict__ out16, int D) {
    __shared__ float sred[16];
    __shared__ float smean, srstd;
    size_t row = blockIdx.x;
    const float* xr = x + row * (size_t)D;
    float s = 0.f, s2 = 0.f;
    for (int i = threadIdx.x; i < D; i += 256) { float v = xr[i]; s += v; s2 += v * v; }
#pragma unroll
    for (int o = 16; o; o >>= 1) { s += __shfl_xor(s, o, 32); s2 += __shfl_xor(s2, o, 32); }
    int wv = threadIdx.x >> 5, ln = threadIdx.x & 31;
    if (ln == 0) { sred[wv] = s; sred[8 + wv] = s2; }
    __syncthreads();
    if (threadIdx.x == 0) {
        float ts = 0.f, ts2 = 0.f;
        for (int i = 0; i < 8; ++i) { ts += sred[i]; ts2 += sred[8 + i]; }
        float m = ts / D;
        smean = m;
        srstd = rsqrtf(ts2 / D - m * m + 1e-5f);
    }
    __syncthreads();
    float m = smean, rs = srstd;
    for (int i = threadIdx.x; i < D; i += 256) {
        float v = (xr[i] - m) * rs * g[i] + b[i];
        if (out32) out32[row * (size_t)D + i] = v;
        if (out16) out16[row * (size_t)D + i] = (_Float16)v;
    }
}

// ============ top-k(32) softmax + head-mean scatter ============
__global__ __launch_bounds__(256) void topk_kernel(const float* __restrict__ scores,
                                                   const unsigned char* __restrict__ vmask,
                                                   float* __restrict__ am) {
    int w    = blockIdx.x * 8 + (threadIdx.x >> 5);
    int lane = threadIdx.x & 31;
    int t = w & (T_ - 1);
    int h = (w >> 9) & (H_ - 1);
    int b = w >> 11;
    const float* srow = scores + (((size_t)(b * H_ + h)) * T_ + t) * G_;

    float v[18];
#pragma unroll
    for (int i = 0; i < 18; ++i) {
        int g = lane + i * 32;
        float s = srow[g];
        if (!vmask[b * G_ + g]) s = -3.0e38f;
        v[i] = s;
    }
    float selv = 0.f; int selg = 0;
    for (int it = 0; it < KTOP_; ++it) {
        float bv = -3.2e38f; int bi = 0;
#pragma unroll
        for (int i = 0; i < 18; ++i) if (v[i] > bv) { bv = v[i]; bi = i; }
        int bg = lane + bi * 32;
#pragma unroll
        for (int o = 16; o; o >>= 1) {
            float ov = __shfl_xor(bv, o, 32);
            int   og = __shfl_xor(bg, o, 32);
            if (ov > bv || (ov == bv && og < bg)) { bv = ov; bg = og; }
        }
        if ((bg & 31) == lane) v[bg >> 5] = -3.2e38f;
        if (lane == it) { selv = bv; selg = bg; }
    }
    float m = __shfl(selv, 0, 32);
    float e = __expf(selv - m);
    float ssum = e;
#pragma unroll
    for (int o = 16; o; o >>= 1) ssum += __shfl_xor(ssum, o, 32);
    float p = e / ssum * (1.0f / H_);
    atomicAdd(&am[((size_t)b * T_ + t) * G_ + selg], p);
}

// ============ S4D: ZOH discretization ============
__global__ __launch_bounds__(256) void s4d_setup(const float* __restrict__ log_dt,
                                                 const float* __restrict__ A_re,
                                                 const float* __restrict__ A_im,
                                                 const float* __restrict__ C_re,
                                                 const float* __restrict__ C_im,
                                                 float* lr, float* li, float* cr, float* ci) {
    int idx = blockIdx.x * 256 + threadIdx.x;
    if (idx >= DT_ * NST_) return;
    int d = idx >> 6;
    float dt = __expf(log_dt[d]);
    float ar = A_re[idx], ai = A_im[idx];
    float xr = dt * ar, xi = dt * ai;
    float er = __expf(xr);
    float Lr = er * cosf(xi), Li = er * sinf(xi);
    float nr = Lr - 1.0f, ni = Li;
    float den = ar * ar + ai * ai;
    float tr = (nr * ar + ni * ai) / den;
    float ti = (ni * ar - nr * ai) / den;
    float Cr = C_re[idx], Ci = C_im[idx];
    cr[idx] = Cr * tr - Ci * ti;
    ci[idx] = Cr * ti + Ci * tr;
    lr[idx] = Lr; li[idx] = Li;
}

// ============ S4D: recurrent scan, one wave per (b,d) ============
__global__ __launch_bounds__(256) void s4d_scan(const float* __restrict__ xin,
                                                const float* __restrict__ lr,
                                                const float* __restrict__ li,
                                                const float* __restrict__ cr,
                                                const float* __restrict__ ci,
                                                const float* __restrict__ dskip,
                                                float* __restrict__ yout) {
    int w    = blockIdx.x * 8 + (threadIdx.x >> 5);
    int lane = threadIdx.x & 31;
    int b = w >> 10;
    int d = w & (DT_ - 1);
    int n0 = d * NST_ + lane, n1 = n0 + 32;
    float l0r = lr[n0], l0i = li[n0], l1r = lr[n1], l1i = li[n1];
    float c0r = cr[n0], c0i = ci[n0], c1r = cr[n1], c1i = ci[n1];
    float ds = dskip[d];
    float s0r = 0.f, s0i = 0.f, s1r = 0.f, s1i = 0.f;
    const float* xp = xin + ((size_t)b * T_) * DT_ + d;
    float*       yp = yout + ((size_t)b * T_) * DT_ + d;
    for (int t = 0; t < T_; ++t) {
        float x = xp[(size_t)t * DT_];
        float a0 = l0r * s0r - l0i * s0i + x;
        float b0 = l0r * s0i + l0i * s0r;
        s0r = a0; s0i = b0;
        float a1 = l1r * s1r - l1i * s1i + x;
        float b1 = l1r * s1i + l1i * s1r;
        s1r = a1; s1i = b1;
        float contrib = c0r * s0r - c0i * s0i + c1r * s1r - c1i * s1i;
#pragma unroll
        for (int o = 16; o; o >>= 1) contrib += __shfl_xor(contrib, o, 32);
        if (lane == 0) yp[(size_t)t * DT_] = 2.0f * contrib + ds * x;
    }
}

// ================= elementwise kernels =================
__global__ __launch_bounds__(256) void film_apply(const float* __restrict__ src,
                                                  const float* __restrict__ film,
                                                  const float* __restrict__ mgp,
                                                  float* __restrict__ dst, size_t n) {
    size_t i = (size_t)blockIdx.x * 256 + threadIdx.x;
    if (i >= n) return;
    size_t row = i >> 10; int col = (int)(i & 1023);
    float m  = mgp[0];
    float sc = film[row * 2048 + col];
    float sh = film[row * 2048 + 1024 + col];
    dst[i] = src[i] * (1.0f + m * sc) + m * sh;
}

__global__ __launch_bounds__(256) void film_apply_res(const float* __restrict__ yin,
                                                      const float* __restrict__ film,
                                                      const float* __restrict__ mgp,
                                                      const float* __restrict__ xres,
                                                      float* __restrict__ dst, size_t n) {
    size_t i = (size_t)blockIdx.x * 256 + threadIdx.x;
    if (i >= n) return;
    size_t row = i >> 10; int col = (int)(i & 1023);
    float m  = mgp[0];
    float sc = film[row * 2048 + col];
    float sh = film[row * 2048 + 1024 + col];
    dst[i] = xres[i] + (yin[i] * (1.0f + m * sc) + m * sh);
}

// ================= host side =================
static inline void gemm_big(hipStream_t st, const _Float16* A, int lda, long long sAb,
                            const _Float16* Bt, int ldb, long long sBb,
                            float* C, _Float16* Ch, _Float16* Cht, int ldc, int ldct, int RB,
                            long long sCb, long long sCtb, int M, int N, int K, int batch,
                            float alpha, int mode, const float* bias, const float* resid,
                            const unsigned char* rmask, long long sRm) {
    GPH p{A, Bt, C, Ch, Cht, bias, resid, rmask, M, N, K, lda, ldb, ldc, ldct, RB, mode,
          sAb, sBb, sCb, sCtb, sRm, alpha};
    dim3 g(N / 128, M / 128, batch), blk(256);
    gemm_h<128, 128, 4, 2, 2, 4, 1><<<g, blk, 0, st>>>(p);
}

static inline void gemm_small(hipStream_t st, const _Float16* A, int lda, long long sAb,
                              const _Float16* Bt, int ldb, long long sBb,
                              float* C, int ldc, long long sCb,
                              int M, int N, int K, int batch, float alpha) {
    GPH p{A, Bt, C, nullptr, nullptr, nullptr, nullptr, nullptr, M, N, K, lda, ldb, ldc, 0, 1, 0,
          sAb, sBb, sCb, 0, 0, alpha};
    dim3 g(N / 64, M / 64, batch), blk(128);
    gemm_h<64, 64, 2, 2, 2, 2, 0><<<g, blk, 0, st>>>(p);
}

extern "C" void kernel_launch(void* const* d_in, const int* in_sizes, int n_in,
                              void* d_out, int out_size, void* d_ws, size_t ws_size,
                              hipStream_t stream) {
    (void)in_sizes; (void)n_in; (void)out_size; (void)ws_size;

    const float* x_text   = (const float*)d_in[0];
    const float* x_vision = (const float*)d_in[1];
    const unsigned char* text_mask   = (const unsigned char*)d_in[2];
    const unsigned char* vision_mask = (const unsigned char*)d_in[3];
    const float* Wq = (const float*)d_in[4];
    const float* Wk = (const float*)d_in[5];
    const float* Wv = (const float*)d_in[6];
    const float* W_film_in  = (const float*)d_in[7];
    const float* b_film_in  = (const float*)d_in[8];
    const float* W_film_out = (const float*)d_in[9];
    const float* b_film_out = (const float*)d_in[10];
    const float* g_norm_in  = (const float*)d_in[11];
    const float* b_norm_in  = (const float*)d_in[12];
    const float* g_norm_out = (const float*)d_in[13];
    const float* b_norm_out = (const float*)d_in[14];
    const float* g_ffn = (const float*)d_in[15];
    const float* b_ffn = (const float*)d_in[16];
    const float* W1 = (const float*)d_in[17];
    const float* b1 = (const float*)d_in[18];
    const float* W2 = (const float*)d_in[19];
    const float* b2 = (const float*)d_in[20];
    const float* log_dt = (const float*)d_in[21];
    const float* A_re = (const float*)d_in[22];
    const float* A_im = (const float*)d_in[23];
    const float* C_re = (const float*)d_in[24];
    const float* C_im = (const float*)d_in[25];
    const float* D_skip = (const float*)d_in[26];
    const float* mod_gate = (const float*)d_in[27];

    // ---- workspace bump allocator (256B aligned) ----
    char* wsb = (char*)d_ws;
    size_t off = 0;
    auto ALO = [&](size_t bytes) -> void* {
        void* p = wsb + off;
        off += (bytes + 255) & ~(size_t)255;
        return p;
    };
    _Float16* xt_h   = (_Float16*)ALO((size_t)B_ * T_ * DT_ * 2);
    _Float16* xv_h   = (_Float16*)ALO((size_t)B_ * G_ * DV_ * 2);
    _Float16* WqT    = (_Float16*)ALO((size_t)DT_ * INNER_ * 2);
    _Float16* WkT    = (_Float16*)ALO((size_t)DV_ * INNER_ * 2);
    _Float16* WvT    = (_Float16*)ALO((size_t)DV_ * DT_ * 2);
    _Float16* WfinT  = (_Float16*)ALO((size_t)DT_ * 2 * DT_ * 2);
    _Float16* WfoutT = (_Float16*)ALO((size_t)DT_ * 2 * DT_ * 2);
    _Float16* W1T    = (_Float16*)ALO((size_t)DT_ * HID_ * 2);
    _Float16* W2T    = (_Float16*)ALO((size_t)HID_ * DT_ * 2);
    _Float16* Q_h    = (_Float16*)ALO((size_t)B_ * T_ * INNER_ * 2);
    _Float16* K_h    = (_Float16*)ALO((size_t)B_ * G_ * INNER_ * 2);
    _Float16* Vt_h   = (_Float16*)ALO((size_t)B_ * DT_ * G_ * 2);  // per-batch V^T (DT x G)
    _Float16* Am_h   = (_Float16*)ALO((size_t)B_ * T_ * G_ * 2);
    _Float16* Ct_h   = (_Float16*)ALO((size_t)B_ * T_ * DT_ * 2);
    _Float16* Hln_h  = (_Float16*)ALO((size_t)B_ * T_ * DT_ * 2);
    _Float16* H1_h   = (_Float16*)ALO((size_t)B_ * T_ * 2048 * 2);
    float* Sc   = (float*)ALO((size_t)B_ * H_ * T_ * G_ * 4);
    float* Am   = (float*)ALO((size_t)B_ * T_ * G_ * 4);
    float* Xln  = (float*)ALO((size_t)B_ * T_ * DT_ * 4);
    float* Film = (float*)ALO((size_t)B_ * T_ * 2 * DT_ * 4);
    float* Yb   = (float*)ALO((size_t)B_ * T_ * DT_ * 4);
    float* LamR = (float*)ALO((size_t)DT_ * NST_ * 4);
    float* LamI = (float*)ALO((size_t)DT_ * NST_ * 4);
    float* DcR  = (float*)ALO((size_t)DT_ * NST_ * 4);
    float* DcI  = (float*)ALO((size_t)DT_ * NST_ * 4);

    const size_t nTD = (size_t)B_ * T_ * DT_;
    const int ewGrid = (int)((nTD + 255) / 256);

    // ---- 0) f16 conversions (activations) + weight transposes ----
    cvt16<<<(int)((nTD + 255) / 256), 256, 0, stream>>>(x_text, xt_h, nTD);
    cvt16<<<(int)(((size_t)B_ * G_ * DV_ + 255) / 256), 256, 0, stream>>>(x_vision, xv_h, (size_t)B_ * G_ * DV_);
    transpose_cvt<<<dim3(INNER_ / 32, DT_ / 32), 256, 0, stream>>>(Wq, WqT, DT_, INNER_);
    transpose_cvt<<<dim3(INNER_ / 32, DV_ / 32), 256, 0, stream>>>(Wk, WkT, DV_, INNER_);
    transpose_cvt<<<dim3(DT_ / 32, DV_ / 32), 256, 0, stream>>>(Wv, WvT, DV_, DT_);
    transpose_cvt<<<dim3(2 * DT_ / 32, DT_ / 32), 256, 0, stream>>>(W_film_in, WfinT, DT_, 2 * DT_);
    transpose_cvt<<<dim3(2 * DT_ / 32, DT_ / 32), 256, 0, stream>>>(W_film_out, WfoutT, DT_, 2 * DT_);
    transpose_cvt<<<dim3(HID_ / 32, DT_ / 32), 256, 0, stream>>>(W1, W1T, DT_, HID_);
    transpose_cvt<<<dim3(DT_ / 32, HID_ / 32), 256, 0, stream>>>(W2, W2T, HID_, DT_);

    // ---- 1) projections ----
    gemm_big(stream, xt_h, DT_, 0, WqT, DT_, 0, nullptr, Q_h, nullptr, INNER_, 0, 1, 0, 0,
             B_ * T_, INNER_, DT_, 1, 1.f, 0, nullptr, nullptr, nullptr, 0);
    gemm_big(stream, xv_h, DV_, 0, WkT, DV_, 0, nullptr, K_h, nullptr, INNER_, 0, 1, 0, 0,
             B_ * G_, INNER_, DV_, 1, 1.f, 0, nullptr, nullptr, nullptr, 0);
    // V written transposed per batch: Vt_h[b][d][g]
    gemm_big(stream, xv_h, DV_, 0, WvT, DV_, 0, nullptr, nullptr, Vt_h, DT_, G_, G_,
             0, (long long)DT_ * G_, B_ * G_, DT_, DV_, 1, 1.f, 0, nullptr, nullptr, nullptr, 0);

    // ---- 2) scores = QK^T/sqrt(DH), per head, batched over b ----
    for (int h = 0; h < H_; ++h) {
        gemm_small(stream, Q_h + h * DH_, INNER_, (long long)T_ * INNER_,
                   K_h + h * DH_, INNER_, (long long)G_ * INNER_,
                   Sc + (size_t)h * T_ * G_, G_, (long long)H_ * T_ * G_,
                   T_, G_, DH_, B_, 0.17677669529663689f);
    }

    // ---- 3) top-k softmax -> attn_mean ----
    (void)hipMemsetAsync(Am, 0, (size_t)B_ * T_ * G_ * sizeof(float), stream);
    topk_kernel<<<(B_ * H_ * T_) / 8, 256, 0, stream>>>(Sc, vision_mask, Am);
    cvt16<<<(int)(((size_t)B_ * T_ * G_ + 255) / 256), 256, 0, stream>>>(Am, Am_h, (size_t)B_ * T_ * G_);

    // ---- 4) c_t = attn_mean @ V (batched), text-mask fused into epilogue ----
    gemm_big(stream, Am_h, G_, (long long)T_ * G_,
             Vt_h, G_, (long long)DT_ * G_,
             nullptr, Ct_h, nullptr, DT_, 0, 1, (long long)T_ * DT_, 0,
             T_, DT_, G_, B_, 1.f, 0, nullptr, nullptr, text_mask, T_);

    // ---- 5) x = FiLM_in(LN(x_text)) ----
    ln_kernel<<<B_ * T_, 256, 0, stream>>>(x_text, g_norm_in, b_norm_in, Xln, nullptr, DT_);
    gemm_big(stream, Ct_h, DT_, 0, WfinT, DT_, 0, Film, nullptr, nullptr, 2 * DT_, 0, 1, 0, 0,
             B_ * T_, 2 * DT_, DT_, 1, 1.f, 1, b_film_in, nullptr, nullptr, 0);
    film_apply<<<ewGrid, 256, 0, stream>>>(Xln, Film, mod_gate, Xln, nTD);

    // ---- 6) S4D recurrence (== the reference's FFT causal conv) ----
    s4d_setup<<<(DT_ * NST_) / 256, 256, 0, stream>>>(log_dt, A_re, A_im, C_re, C_im,
                                                      LamR, LamI, DcR, DcI);
    s4d_scan<<<(B_ * DT_) / 8, 256, 0, stream>>>(Xln, LamR, LamI, DcR, DcI, D_skip, Yb);

    // ---- 7) xt = x_text + FiLM_out(y) ----
    gemm_big(stream, Ct_h, DT_, 0, WfoutT, DT_, 0, Film, nullptr, nullptr, 2 * DT_, 0, 1, 0, 0,
             B_ * T_, 2 * DT_, DT_, 1, 1.f, 1, b_film_out, nullptr, nullptr, 0);
    film_apply_res<<<ewGrid, 256, 0, stream>>>(Yb, Film, mod_gate, x_text, Yb, nTD);

    // ---- 8) FFN (two 2048-wide chunks), residual accumulated into Yb ----
    ln_kernel<<<B_ * T_, 256, 0, stream>>>(Yb, g_ffn, b_ffn, nullptr, Hln_h, DT_);
    for (int c = 0; c < 2; ++c) {
        gemm_big(stream, Hln_h, DT_, 0, W1T + (size_t)c * 2048 * DT_, DT_, 0,
                 nullptr, H1_h, nullptr, 2048, 0, 1, 0, 0,
                 B_ * T_, 2048, DT_, 1, 1.f, 2, b1 + c * 2048, nullptr, nullptr, 0);
        gemm_big(stream, H1_h, 2048, 0, W2T + (size_t)c * 2048, HID_, 0,
                 Yb, nullptr, nullptr, DT_, 0, 1, 0, 0,
                 B_ * T_, DT_, 2048, 1, 1.f, 3, (c == 0 ? b2 : nullptr), Yb, nullptr, 0);
    }

    // ---- 9) final LayerNorm -> d_out ----
    ln_kernel<<<B_ * T_, 256, 0, stream>>>(Yb, g_norm_out, b_norm_out, (float*)d_out, nullptr, DT_);
}
